// RNN_3556232921204
// MI455X (gfx1250) — matrix operand order repaired and checked by reading, not compiled, not run on hardware
//
#include <hip/hip_runtime.h>
#include <cstdint>
#include <cstddef>

// MI455X / gfx1250, wave32. WMMA f16 16x16x32 with f32 accum.
//
// Roofline: ig-GEMM phase is HBM-bound (x = 381MB read once, ig = 192MiB f16
// written then re-read once by the scan; ~25-35us total at 23.3 TB/s), so f16
// storage + WMMA is the right operating point. The scan is latency-bound
// (T=2048 serial steps, parallelism capped at 16 M-tiles x 12 N-tiles), so we
// run 16 persistent workgroups x 12 waves with W_hh fragments pinned in VGPRs,
// h staged in LDS, and the next timestep's ig slice streamed a step ahead via
// GLOBAL_LOAD_ASYNC_TO_LDS_B128 (ASYNCcnt double-buffer).

typedef __attribute__((ext_vector_type(16))) _Float16 v16h;
typedef __attribute__((ext_vector_type(8)))  _Float16 v8h;
typedef __attribute__((ext_vector_type(8)))  float    v8f;

#define B_   256
#define T_   2048
#define IN_  182
#define KP   192      // K padded to multiple of 32
#define H_   64
#define G3   192      // 3*H
#define OUT_ 2

__device__ inline v16h join8(v8h lo, v8h hi) {
  v16h f;
#pragma unroll
  for (int q = 0; q < 8; ++q) { f[q] = lo[q]; f[q + 8] = hi[q]; }
  return f;
}

__device__ inline float sigmoidf_(float v) { return 1.f / (1.f + __expf(-v)); }

// Async copy of 16B/lane global -> LDS, tracked by ASYNCcnt.
__device__ inline void async_b128_to_lds(uint32_t lds_byte_off, uint64_t gaddr) {
  asm volatile("global_load_async_to_lds_b128 %0, %1, off"
               :: "v"(lds_byte_off), "v"(gaddr) : "memory");
}

template <int N>
__device__ inline void wait_asynccnt() {
#if defined(__has_builtin)
#if __has_builtin(__builtin_amdgcn_s_wait_asynccnt)
  __builtin_amdgcn_s_wait_asynccnt(N);
#else
  asm volatile("s_wait_asynccnt %0" :: "i"(N) : "memory");
#endif
#else
  asm volatile("s_wait_asynccnt %0" :: "i"(N) : "memory");
#endif
}

// ---------------------------------------------------------------------------
// Phase 0: convert weights to f16, zero-pad K of W_ih from 182 -> 192.
// ---------------------------------------------------------------------------
__global__ void prep_weights(const float* __restrict__ wih,
                             const float* __restrict__ whh,
                             _Float16* __restrict__ w16,
                             _Float16* __restrict__ whh16) {
  int i = blockIdx.x * blockDim.x + threadIdx.x;
  int stride = gridDim.x * blockDim.x;
  for (int idx = i; idx < G3 * KP; idx += stride) {
    int n = idx / KP, k = idx % KP;
    w16[idx] = (k < IN_) ? (_Float16)wih[n * IN_ + k] : (_Float16)0.f;
  }
  for (int idx = i; idx < G3 * H_; idx += stride)
    whh16[idx] = (_Float16)whh[idx];
}

// ---------------------------------------------------------------------------
// Phase 1: ig[b][t][n] = x[b,t,:] @ W_ih^T + bias, stored f16.
// Tile = one batch row b, 16 timesteps (M), 12 waves cover N=192.
// W_ih B-fragments live in VGPRs across the whole M loop.
// ---------------------------------------------------------------------------
#define XS_W 200   // padded LDS row (halves): 36m mod 64 banks -> conflict-free
__global__ __launch_bounds__(384) void ig_gemm(const float* __restrict__ x,
                                               const _Float16* __restrict__ w16,
                                               const float* __restrict__ bias,
                                               _Float16* __restrict__ ig) {
  __shared__ __align__(16) _Float16 xs[16 * XS_W];
  const int tid  = threadIdx.x;
  const int wave = tid >> 5;          // 0..11 -> N tile
  const int lane = tid & 31;
  const int hi   = lane >> 4;
  const int n0   = wave * 16;
  const int ncol = n0 + (lane & 15);

  // Resident B fragments: lane holds column ncol; lanes0-15 K=0-7,16-23 of the
  // 32-wide K step, lanes16-31 K=8-15,24-31 -> two 8-half (16B) chunks.
  v16h bfrag[6];
#pragma unroll
  for (int s = 0; s < 6; ++s)
    bfrag[s] = join8(*(const v8h*)(w16 + (size_t)ncol * KP + s * 32 + hi * 8),
                     *(const v8h*)(w16 + (size_t)ncol * KP + s * 32 + 16 + hi * 8));
  const float bv = bias[ncol];

  const int m   = lane & 15;
  const int ka0 = hi ? 8 : 0;
  const int ntiles = B_ * (T_ / 16);

  for (int tile = blockIdx.x; tile < ntiles; tile += gridDim.x) {
    const int b  = tile >> 7;              // 128 tiles per b
    const int t0 = (tile & 127) << 4;
    __syncthreads();                       // xs reads of prev tile done
    for (int idx = tid; idx < 16 * XS_W; idx += 384) {
      int r = idx / XS_W, k = idx % XS_W;
      float vv = (k < IN_) ? x[((size_t)b * T_ + (t0 + r)) * IN_ + k] : 0.f;
      xs[idx] = (_Float16)vv;
    }
    __syncthreads();

    v8f c;
#pragma unroll
    for (int q = 0; q < 8; ++q) c[q] = bv;   // bias depends on N only

#pragma unroll
    for (int s = 0; s < 6; ++s) {
      const _Float16* p = xs + m * XS_W + s * 32 + ka0;
      v16h a = join8(*(const v8h*)p, *(const v8h*)(p + 16));
      c = __builtin_amdgcn_wmma_f32_16x16x32_f16(false, a, false, bfrag[s],
                                                 (short)0, c, false, false);
    }
    // D element (Mrow = v + 8*hi -> timestep, N = ncol)
#pragma unroll
    for (int v = 0; v < 8; ++v) {
      int t = t0 + v + 8 * hi;
      ig[((size_t)b * T_ + t) * G3 + ncol] = (_Float16)c[v];
    }
  }
}

// ---------------------------------------------------------------------------
// Phase 2: persistent GRU scan. 16 WGs (16 batch rows each) x 12 waves.
// Per step per wave: 2x v_wmma (h @ W_hh^T tile), 1 async global->LDS copy of
// the NEXT step's ig tile (double buffered), LDS gate exchange, elementwise
// update, f16 restage of h.
// ---------------------------------------------------------------------------
#define PRE_W 196   // padded f32 row to break 2-way bank conflicts
#define H16_W 72    // padded f16 row (144B): 36m mod 64 banks
__global__ __launch_bounds__(384) void gru_scan(const _Float16* __restrict__ ig,
                                                const _Float16* __restrict__ whh16,
                                                const float* __restrict__ bias_n,
                                                const float* __restrict__ w_out,
                                                const float* __restrict__ out_bias,
                                                float* __restrict__ out) {
  __shared__ float preA[16][PRE_W];                     // hg pre-activations
  __shared__ __align__(16) _Float16 igb[2][16][G3];     // async double buffer
  __shared__ float h32[16][H_];
  __shared__ __align__(16) _Float16 h16[16][H16_W];
  __shared__ float bn[H_];

  const int tid  = threadIdx.x;
  const int wave = tid >> 5;
  const int lane = tid & 31;
  const int hi   = lane >> 4;
  const int n0   = wave * 16;
  const int ncol = n0 + (lane & 15);
  const int b0   = blockIdx.x * 16;

  for (int i = tid; i < 16 * H16_W; i += 384) h16[i / H16_W][i % H16_W] = (_Float16)0.f;
  for (int i = tid; i < 16 * H_; i += 384)   h32[i >> 6][i & 63] = 0.f;
  for (int i = tid; i < H_; i += 384)        bn[i] = bias_n[i];

  // Resident W_hh^T B fragments (K=64 -> 2 steps of 32)
  v16h bfrag[2];
#pragma unroll
  for (int s = 0; s < 2; ++s)
    bfrag[s] = join8(*(const v8h*)(whh16 + (size_t)ncol * H_ + s * 32 + hi * 8),
                     *(const v8h*)(whh16 + (size_t)ncol * H_ + s * 32 + 16 + hi * 8));

  // Async ig stream: lane copies 8 halves (16B) of row (lane&15), half-block
  // (lane>>4) of this wave's n-tile. Address advances 384B per timestep.
  const int arow = lane & 15;
  const int acol = n0 + (hi << 3);                      // n0 + {0,8}
  uint64_t ga = (uint64_t)(uintptr_t)ig +
                (((uint64_t)(b0 + arow) * T_) * G3 + acol) * sizeof(_Float16);
  const uint32_t la0 = (uint32_t)(uintptr_t)&igb[0][arow][acol];
  const uint32_t la1 = (uint32_t)(uintptr_t)&igb[1][arow][acol];

  async_b128_to_lds(la0, ga);                           // t = 0 -> buffer 0
  __syncthreads();

  const int m   = lane & 15;
  const int ka0 = hi ? 8 : 0;

  for (int t = 0; t < T_; ++t) {
    // hg tile = h(16x64,f16) @ W_hh^T tile
    v8f c = {};
#pragma unroll
    for (int s = 0; s < 2; ++s) {
      const _Float16* p = &h16[m][s * 32 + ka0];
      v16h a = join8(*(const v8h*)p, *(const v8h*)(p + 16));
      c = __builtin_amdgcn_wmma_f32_16x16x32_f16(false, a, false, bfrag[s],
                                                 (short)0, c, false, false);
    }
    // kick off next step's ig tile while this step computes
    if (t + 1 < T_) {
      ga += G3 * sizeof(_Float16);
      async_b128_to_lds(((t + 1) & 1) ? la1 : la0, ga);
      wait_asynccnt<1>();             // older copy (this step's) is complete
    } else {
      wait_asynccnt<0>();
    }
#pragma unroll
    for (int v = 0; v < 8; ++v) preA[v + 8 * hi][ncol] = c[v];
    __syncthreads();

    // gates + state update: 1024 elems over 384 threads
    const _Float16* igt = &igb[t & 1][0][0];
    for (int j = tid; j < 16 * H_; j += 384) {
      int row = j >> 6, col = j & 63;
      float igr = (float)igt[row * G3 + col];
      float igz = (float)igt[row * G3 + col + H_];
      float ign = (float)igt[row * G3 + col + 2 * H_];
      float r  = sigmoidf_(igr + preA[row][col]);
      float z  = sigmoidf_(igz + preA[row][col + H_]);
      float nn = tanhf(ign + r * (preA[row][col + 2 * H_] + bn[col]));
      float hn = nn + z * (h32[row][col] - nn);
      h32[row][col] = hn;
      h16[row][col] = (_Float16)hn;
    }
    __syncthreads();
  }

  // readout: sigmoid(h @ w_out^T + out_bias), 16x2 per WG
  if (tid < 32) {
    int row = tid & 15, o = tid >> 4;
    float acc = out_bias[o];
#pragma unroll 8
    for (int k = 0; k < H_; ++k) acc += h32[row][k] * w_out[o * H_ + k];
    out[(b0 + row) * OUT_ + o] = sigmoidf_(acc);
  }
}

// ---------------------------------------------------------------------------
extern "C" void kernel_launch(void* const* d_in, const int* in_sizes, int n_in,
                              void* d_out, int out_size, void* d_ws, size_t ws_size,
                              hipStream_t stream) {
  (void)in_sizes; (void)n_in; (void)out_size; (void)ws_size;
  const float* x        = (const float*)d_in[0];
  const float* wih      = (const float*)d_in[1];
  const float* whh      = (const float*)d_in[2];
  const float* bias     = (const float*)d_in[3];
  const float* bias_n   = (const float*)d_in[4];
  const float* w_out    = (const float*)d_in[5];
  const float* out_bias = (const float*)d_in[6];
  float* out = (float*)d_out;

  // ws layout: [0,96KB) f16 weights, [128KB, 128KB+192MiB) f16 ig buffer
  _Float16* w16   = (_Float16*)d_ws;                    // 192*192 halves
  _Float16* whh16 = w16 + G3 * KP;                      // 192*64 halves
  _Float16* ig    = (_Float16*)((char*)d_ws + (128 << 10));

  prep_weights<<<96, 256, 0, stream>>>(wih, whh, w16, whh16);
  ig_gemm<<<2048, 384, 0, stream>>>(x, w16, bias, ig);
  gru_scan<<<16, 384, 0, stream>>>(ig, whh16, bias_n, w_out, out_bias, out);
}
